// GNSSMessagePassing_14121852469802
// MI455X (gfx1250) — matrix operand order repaired and checked by reading, not compiled, not run on hardware
//
#include <hip/hip_runtime.h>
#include <math.h>

#define HID   128
#define EDIM  64
#define K1    (2*HID+EDIM)   // 320
#define KU1   (2*HID)        // 256

typedef __attribute__((ext_vector_type(16))) _Float16 v16h;
typedef __attribute__((ext_vector_type(8)))  _Float16 v8h;
typedef __attribute__((ext_vector_type(8)))  float    v8f;

// tanh-form GELU with cheap tanh: tanh(u) = 1 - 2/(1+exp(2u))  (stable at +/-inf)
__device__ __forceinline__ float gelu_tanh(float x) {
    float u = 0.7978845608028654f * (x + 0.044715f * x * x * x);
    float e = __expf(2.0f * u);
    float t = 1.0f - 2.0f / (1.0f + e);
    return 0.5f * x * (1.0f + t);
}

// ---- A fragment (16x32 f16). Row-major buffer, row stride ld (halfs).
// lane L holds row rowBase+(L&15). lanes<16: K in {0..7,16..23}; lanes>=16: K in {8..15,24..31}.
__device__ __forceinline__ v16h load_fragA(const _Float16* buf, int rowBase, int ld,
                                           int k0, int lane) {
    const _Float16* p = buf + (rowBase + (lane & 15)) * ld + k0 + ((lane >= 16) ? 8 : 0);
    v8h lo = *(const v8h*)(p);
    v8h hi = *(const v8h*)(p + 16);
    return __builtin_shufflevector(lo, hi, 0,1,2,3,4,5,6,7,8,9,10,11,12,13,14,15);
}

// ---- B fragment (32x16 f16). Weights pre-transposed to [N][K] row-major (stride ld = K).
// lane L holds column colBase+(L&15). lanes<16: K in {0..7,16..23}; lanes>=16: K in {8..15,24..31}.
__device__ __forceinline__ v16h load_fragB(const _Float16* wt, int colBase, int ld,
                                           int k0, int lane) {
    const _Float16* p = wt + (colBase + (lane & 15)) * ld + k0 + ((lane >= 16) ? 8 : 0);
    v8h lo = *(const v8h*)(p);
    v8h hi = *(const v8h*)(p + 16);
    return __builtin_shufflevector(lo, hi, 0,1,2,3,4,5,6,7,8,9,10,11,12,13,14,15);
}

// One wave computes a 32xHID (2 M-tiles x 8 N-tiles) output block.
// Each B fragment is loaded once and reused for both M-tiles (halves L2 weight traffic).
template <int KTOT>
__device__ __forceinline__ void wave_gemm2(const _Float16* Abuf, int ldA, int rowBase,
                                           const _Float16* Wt, const float* __restrict__ bias,
                                           v8f acc[2][8], int lane) {
    int col = lane & 15;
#pragma unroll
    for (int t = 0; t < 8; ++t) {
        float bv = bias[t * 16 + col];
#pragma unroll
        for (int m = 0; m < 2; ++m)
#pragma unroll
            for (int r = 0; r < 8; ++r) acc[m][t][r] = bv;
    }
    for (int k0 = 0; k0 < KTOT; k0 += 32) {
        v16h a0 = load_fragA(Abuf, rowBase,      ldA, k0, lane);
        v16h a1 = load_fragA(Abuf, rowBase + 16, ldA, k0, lane);
#pragma unroll
        for (int t = 0; t < 8; ++t) {
            v16h b = load_fragB(Wt, t * 16, KTOT, k0, lane);
            acc[0][t] = __builtin_amdgcn_wmma_f32_16x16x32_f16(
                false, a0, false, b, (short)0, acc[0][t], false, false);
            acc[1][t] = __builtin_amdgcn_wmma_f32_16x16x32_f16(
                false, a1, false, b, (short)0, acc[1][t], false, false);
        }
    }
}

// Store wave's 32x128 accumulator block to row-major f16 (optionally through GELU).
// D layout: VGPR r holds M = r + (lane>=16 ? 8 : 0), N = lane&15 (+16*t).
__device__ __forceinline__ void store_acc_f16(_Float16* dst, int rowBase, int ld,
                                              const v8f acc[2][8], int lane, bool act) {
    int col  = lane & 15;
    int roff = (lane >> 4) * 8;
#pragma unroll
    for (int m = 0; m < 2; ++m) {
#pragma unroll
        for (int t = 0; t < 8; ++t) {
#pragma unroll
            for (int r = 0; r < 8; ++r) {
                float v = acc[m][t][r];
                if (act) v = gelu_tanh(v);
                dst[(rowBase + m * 16 + roff + r) * ld + t * 16 + col] = (_Float16)v;
            }
        }
    }
}

// ---------------------------------------------------------------- prep kernels
__global__ void zero_f32_kernel(float* p, long n) {
    long i = (long)blockIdx.x * blockDim.x + threadIdx.x;
    long stride = (long)gridDim.x * blockDim.x;
    for (; i < n; i += stride) p[i] = 0.0f;
}

// dst[n*K + k] = (f16) src[k*N + n]   (transpose + f32->f16)
__global__ void cvt_transpose_kernel(const float* __restrict__ src, _Float16* __restrict__ dst,
                                     int K, int N) {
    int tot = K * N;
    for (int idx = blockIdx.x * blockDim.x + threadIdx.x; idx < tot;
         idx += gridDim.x * blockDim.x) {
        int k = idx / N, n = idx - k * N;
        dst[n * K + k] = (_Float16)src[idx];
    }
}

// ---------------------------------------------------------------- edge message kernel
// 64 edges per block, 2 waves; each wave owns 32 edge rows (2 M-tiles).
__global__ __launch_bounds__(64) void msg_kernel(
    const float* __restrict__ h, const int* __restrict__ src, const int* __restrict__ dst,
    const float* __restrict__ ea,
    const _Float16* __restrict__ W1t, const float* __restrict__ b1,
    const _Float16* __restrict__ W2t, const float* __restrict__ b2,
    const _Float16* __restrict__ W3t, const float* __restrict__ b3,
    float* __restrict__ agg, float* __restrict__ cnt, int E) {
    __shared__ __align__(32) _Float16 sX[64 * K1];   // 40 KB input (also reused as Y2)
    __shared__ __align__(32) _Float16 sY[64 * HID];  // 16 KB
    __shared__ int sS[64], sD[64];

    int tid = threadIdx.x, lane = tid & 31, wave = tid >> 5;
    int e0 = blockIdx.x * 64;

    {
        int e = e0 + tid;
        int s = 0, d = 0;
        if (e < E) {
            s = src[e]; d = dst[e];
            atomicAdd(&cnt[d], 1.0f);
        }
        sS[tid] = s; sD[tid] = d;
    }
    __syncthreads();

    // Stage X = [h[src] | h[dst] | edge_attr] as f16, 64 x 320
    for (int idx = tid; idx < 64 * K1; idx += 64) {
        int m = idx / K1, c = idx - m * K1;
        int e = e0 + m;
        float v = 0.0f;
        if (e < E) {
            if (c < HID)          v = h[(long)sS[m] * HID + c];
            else if (c < 2 * HID) v = h[(long)sD[m] * HID + (c - HID)];
            else                  v = ea[(long)e * EDIM + (c - 2 * HID)];
        }
        sX[idx] = (_Float16)v;
    }
    __syncthreads();

    int rowBase = wave * 32;
    v8f acc[2][8];

    wave_gemm2<K1>(sX, K1, rowBase, W1t, b1, acc, lane);
    store_acc_f16(sY, rowBase, HID, acc, lane, true);
    __syncthreads();                       // all waves done reading sX

    wave_gemm2<HID>(sY, HID, rowBase, W2t, b2, acc, lane);
    _Float16* sY2 = sX;                    // sX free now; reuse as 64x128 buffer
    store_acc_f16(sY2, rowBase, HID, acc, lane, true);
    __syncthreads();

    wave_gemm2<HID>(sY2, HID, rowBase, W3t, b3, acc, lane);

    // Scatter-add messages into agg[dst]
    int col = lane & 15, roff = (lane >> 4) * 8;
#pragma unroll
    for (int mt = 0; mt < 2; ++mt) {
#pragma unroll
        for (int t = 0; t < 8; ++t) {
#pragma unroll
            for (int r = 0; r < 8; ++r) {
                int m = rowBase + mt * 16 + roff + r;
                int e = e0 + m;
                if (e < E) atomicAdd(&agg[(long)sD[m] * HID + t * 16 + col], acc[mt][t][r]);
            }
        }
    }
}

// ---------------------------------------------------------------- node update kernel
// 64 nodes per block, 2 waves; each wave owns 32 node rows (2 M-tiles).
__global__ __launch_bounds__(64) void upd_kernel(
    const float* __restrict__ h, const float* __restrict__ agg, const float* __restrict__ cnt,
    const _Float16* __restrict__ U1t, const float* __restrict__ ub1,
    const _Float16* __restrict__ U2t, const float* __restrict__ ub2,
    const float* __restrict__ gamma, const float* __restrict__ beta,
    float* __restrict__ out, int N) {
    __shared__ __align__(32) _Float16 sU[64 * KU1];  // 32 KB (later aliased as f32 64x128)
    __shared__ __align__(32) _Float16 sY[64 * HID];  // 16 KB

    int tid = threadIdx.x, lane = tid & 31, wave = tid >> 5;
    int n0 = blockIdx.x * 64;

    // Stage U = [h | agg/count] as f16, 64 x 256
    for (int idx = tid; idx < 64 * KU1; idx += 64) {
        int m = idx / KU1, c = idx - m * KU1;
        int i = n0 + m;
        float v = 0.0f;
        if (i < N) {
            if (c < HID) v = h[(long)i * HID + c];
            else         v = agg[(long)i * HID + (c - HID)] / (cnt[i] + 1e-8f);
        }
        sU[idx] = (_Float16)v;
    }
    __syncthreads();

    int rowBase = wave * 32;
    v8f acc[2][8];

    wave_gemm2<KU1>(sU, KU1, rowBase, U1t, ub1, acc, lane);
    store_acc_f16(sY, rowBase, HID, acc, lane, true);
    __syncthreads();                       // all waves done reading sU

    wave_gemm2<HID>(sY, HID, rowBase, U2t, ub2, acc, lane);

    // Residual add into f32 LDS buffer (aliases freed sU region)
    float* sF = (float*)sU;                // 64*128 f32 = 32 KB
    int col = lane & 15, roff = (lane >> 4) * 8;
#pragma unroll
    for (int mt = 0; mt < 2; ++mt) {
#pragma unroll
        for (int t = 0; t < 8; ++t) {
#pragma unroll
            for (int r = 0; r < 8; ++r) {
                int m = rowBase + mt * 16 + roff + r;
                int i = n0 + m;
                float v = acc[mt][t][r];
                if (i < N) v += h[(long)i * HID + t * 16 + col];
                sF[m * HID + t * 16 + col] = v;
            }
        }
    }
    __syncthreads();

    // LayerNorm per row (one thread per node row)
    {
        int m = tid, i = n0 + m;
        if (i < N) {
            float mu = 0.0f, s2 = 0.0f;
            for (int c = 0; c < HID; ++c) {
                float y = sF[m * HID + c];
                mu += y; s2 += y * y;
            }
            mu *= (1.0f / HID);
            s2 = s2 * (1.0f / HID) - mu * mu;
            float inv = rsqrtf(s2 + 1e-5f);
            for (int c = 0; c < HID; ++c) {
                float y = sF[m * HID + c];
                out[(long)i * HID + c] = (y - mu) * inv * gamma[c] + beta[c];
            }
        }
    }
}

// ---------------------------------------------------------------- host launcher
static inline size_t align256(size_t x) { return (x + 255) & ~(size_t)255; }

extern "C" void kernel_launch(void* const* d_in, const int* in_sizes, int n_in,
                              void* d_out, int out_size, void* d_ws, size_t ws_size,
                              hipStream_t stream) {
    const float* h    = (const float*)d_in[0];
    const int*   ei   = (const int*)d_in[1];
    const float* ea   = (const float*)d_in[2];
    const float* mW1  = (const float*)d_in[3];
    const float* mb1  = (const float*)d_in[4];
    const float* mW2  = (const float*)d_in[5];
    const float* mb2  = (const float*)d_in[6];
    const float* mW3  = (const float*)d_in[7];
    const float* mb3  = (const float*)d_in[8];
    const float* uW1  = (const float*)d_in[9];
    const float* ub1  = (const float*)d_in[10];
    const float* uW2  = (const float*)d_in[11];
    const float* ub2  = (const float*)d_in[12];
    const float* gam  = (const float*)d_in[13];
    const float* bet  = (const float*)d_in[14];

    int N = in_sizes[0] / HID;      // 50000
    int E = in_sizes[1] / 2;        // 600000
    const int* src = ei;
    const int* dst = ei + E;

    // Workspace carve-out
    char* ws = (char*)d_ws;
    size_t off = 0;
    float* agg = (float*)(ws + off); off = align256(off + (size_t)N * HID * sizeof(float));
    float* cnt = (float*)(ws + off); off = align256(off + (size_t)N * sizeof(float));
    _Float16* W1t = (_Float16*)(ws + off); off = align256(off + (size_t)K1  * HID * sizeof(_Float16));
    _Float16* W2t = (_Float16*)(ws + off); off = align256(off + (size_t)HID * HID * sizeof(_Float16));
    _Float16* W3t = (_Float16*)(ws + off); off = align256(off + (size_t)HID * HID * sizeof(_Float16));
    _Float16* U1t = (_Float16*)(ws + off); off = align256(off + (size_t)KU1 * HID * sizeof(_Float16));
    _Float16* U2t = (_Float16*)(ws + off); off = align256(off + (size_t)HID * HID * sizeof(_Float16));
    (void)ws_size; (void)n_in;

    // 1) zero accumulators
    long nz = (long)N * HID + N;
    zero_f32_kernel<<<2048, 256, 0, stream>>>(agg, nz);

    // 2) weight transpose + f32->f16
    cvt_transpose_kernel<<<160, 256, 0, stream>>>(mW1, W1t, K1,  HID);
    cvt_transpose_kernel<<<64,  256, 0, stream>>>(mW2, W2t, HID, HID);
    cvt_transpose_kernel<<<64,  256, 0, stream>>>(mW3, W3t, HID, HID);
    cvt_transpose_kernel<<<128, 256, 0, stream>>>(uW1, U1t, KU1, HID);
    cvt_transpose_kernel<<<64,  256, 0, stream>>>(uW2, U2t, HID, HID);

    // 3) edge messages + scatter
    int gridE = (E + 63) / 64;
    msg_kernel<<<gridE, 64, 0, stream>>>(h, src, dst, ea, W1t, mb1, W2t, mb2, W3t, mb3,
                                         agg, cnt, E);

    // 4) node update + LayerNorm
    int gridN = (N + 63) / 64;
    upd_kernel<<<gridN, 64, 0, stream>>>(h, agg, cnt, U1t, ub1, U2t, ub2, gam, bet,
                                         (float*)d_out, N);
}
